// ScaleDotProductAttention_34351148434235
// MI455X (gfx1250) — matrix operand order repaired and checked by reading, not compiled, not run on hardware
//
#include <hip/hip_runtime.h>

// ---------------- CDNA5 WMMA types ----------------
typedef __attribute__((ext_vector_type(16))) __bf16 v16bf;
typedef __attribute__((ext_vector_type(8)))  float  v8f;

#define B_    2
#define H_    16
#define L_    2048
#define D_    128
#define NEGV  (-10000.0f)

#define KC    32            // keys processed per chunk (WMMA K for the PV matmul)
#define QW    16            // query rows per wave
#define WAVES 4
#define QB    (QW * WAVES)  // 64 query rows per workgroup

// LDS pitches (element counts), padded to break bank conflicts
#define QPITCH 136          // Q tile per wave: 16 rows x 128 bf16 (+8 pad)
#define KPITCH 136          // K tile: 32 rows x 128 bf16 (+8 pad)
#define VPITCH 40           // V^T tile: 128 rows x 32 bf16 (+8 pad)
#define PPITCH 40           // P tile per wave: 16 rows x 32 bf16 (+8 pad)
#define MPITCH 36           // mask tile: 64 rows x 32 int (+4 pad)

union Frag {
  unsigned int   d[8];
  unsigned short u[16];
  v16bf          v;
};

__device__ inline unsigned short f2bf(float f) {
  // round-to-nearest-even fp32 -> bf16
  unsigned int x = __builtin_bit_cast(unsigned int, f);
  x += 0x7FFFu + ((x >> 16) & 1u);
  return (unsigned short)(x >> 16);
}

// Issue ALL of one chunk's global traffic (K, V, mask) into registers.
// No waits here: the LDS-staging step at the top of the next iteration is the
// single loadcnt sync point, so the whole HBM round trip overlaps the current
// chunk's WMMA work.
__device__ inline void issue_chunk_loads(const float* __restrict__ Kp,
                                         const float* __restrict__ Vp,
                                         const int*   __restrict__ Mrow,
                                         int kc, int skey, int dseg,
                                         float4 pk[8], float4 pv[8], int4 pm[4]) {
  const float4* kr = (const float4*)(Kp + (size_t)(kc + skey) * D_ + dseg);
  const float4* vr = (const float4*)(Vp + (size_t)(kc + skey) * D_ + dseg);
#pragma unroll
  for (int j = 0; j < 8; ++j) { pk[j] = kr[j]; pv[j] = vr[j]; }
  const int4* mr = (const int4*)(Mrow + kc);
#pragma unroll
  for (int j = 0; j < 4; ++j) pm[j] = mr[j];
}

__global__ __launch_bounds__(128, 1)
void fa_fwd_bf16_wmma(const float* __restrict__ Q,
                      const float* __restrict__ K,
                      const float* __restrict__ V,
                      const int*   __restrict__ Msk,
                      float*       __restrict__ O) {
  __shared__ unsigned short Qls[WAVES * QW * QPITCH]; // Q tiles, row-major bf16
  __shared__ unsigned short Ks[KC * KPITCH];          // K chunk, row-major bf16
  __shared__ unsigned short VT[D_ * VPITCH];          // V chunk, transposed bf16
  __shared__ unsigned short Ps[WAVES * QW * PPITCH];  // per-wave P scratch
  __shared__ int            Ms[QB * MPITCH];          // mask tile (64 q-rows x 32 keys)

  const int tid  = threadIdx.x;
  const int wave = tid >> 5;
  const int lane = tid & 31;
  const int ln16 = lane & 15;
  const int hi   = lane >> 4;          // lane-half: 0 = lanes 0-15, 1 = lanes 16-31

  const int qblk = blockIdx.x * QB;             // first query row of this block
  const int q0   = qblk + wave * QW;            // first query row of this wave
  const int bh   = blockIdx.y;                  // fused batch*head index

  const float* Qp = Q   + (size_t)bh * L_ * D_;
  const float* Kp = K   + (size_t)bh * L_ * D_;
  const float* Vp = V   + (size_t)bh * L_ * D_;
  const int*   Mp = Msk + (size_t)bh * L_ * L_;
  float*       Op = O   + (size_t)bh * L_ * D_;

  // staging geometry: K/V: 4 threads per key, 32 head-dims each
  const int skey = tid >> 2;
  const int dseg = (tid & 3) * 32;
  // mask staging: 2 threads per query row, 16 keys each
  const int mrow = tid >> 1;                    // 0..63 within the block
  const int mcol = (tid & 1) * 16;
  const int* Mrow = Mp + (size_t)(qblk + mrow) * L_ + mcol;

  // ---- prologue: wave's Q tile (16 x 128) -> LDS as bf16 (wave-private region) ----
  {
    const int cb = hi * 64;                    // each lane covers 64 head-dims of row ln16
    const float4* qr = (const float4*)(Qp + (size_t)(q0 + ln16) * D_ + cb);
    unsigned short* qd = &Qls[(wave * QW + ln16) * QPITCH + cb];
#pragma unroll
    for (int j = 0; j < 16; ++j) {
      const float4 qv = qr[j];
      qd[j * 4 + 0] = f2bf(qv.x); qd[j * 4 + 1] = f2bf(qv.y);
      qd[j * 4 + 2] = f2bf(qv.z); qd[j * 4 + 3] = f2bf(qv.w);
    }
  }

  // O accumulators: 8 d-tiles of 16x16 f32 (C layout: vgpr r -> row r + 8*hi, col = lane%16)
  v8f acc[8];
#pragma unroll
  for (int t = 0; t < 8; ++t)
#pragma unroll
    for (int r = 0; r < 8; ++r) acc[t][r] = 0.0f;

  float mrowv[8], lrow[8];
#pragma unroll
  for (int r = 0; r < 8; ++r) { mrowv[r] = -1e30f; lrow[r] = 0.0f; }

  const float scale = 0.08838834764831845f;   // 1/sqrt(128)

  // ---- software pipeline: chunk 0's K/V/mask loads issued before the loop ----
  float4 pk[8], pv[8];
  int4   pm[4];
  issue_chunk_loads(Kp, Vp, Mrow, 0, skey, dseg, pk, pv, pm);

  for (int kc = 0; kc < L_; kc += KC) {
    __syncthreads();   // previous iteration's LDS consumers done before restaging

    // ---- stage prefetched K (row-major), V (transposed), mask into LDS ----
    {
      unsigned short* kd = &Ks[skey * KPITCH + dseg];
#pragma unroll
      for (int j = 0; j < 8; ++j) {
        const float4 kv = pk[j];
        kd[j * 4 + 0] = f2bf(kv.x); kd[j * 4 + 1] = f2bf(kv.y);
        kd[j * 4 + 2] = f2bf(kv.z); kd[j * 4 + 3] = f2bf(kv.w);
        const float4 vv = pv[j];
        VT[(dseg + j * 4 + 0) * VPITCH + skey] = f2bf(vv.x);
        VT[(dseg + j * 4 + 1) * VPITCH + skey] = f2bf(vv.y);
        VT[(dseg + j * 4 + 2) * VPITCH + skey] = f2bf(vv.z);
        VT[(dseg + j * 4 + 3) * VPITCH + skey] = f2bf(vv.w);
      }
      int* md = &Ms[mrow * MPITCH + mcol];
#pragma unroll
      for (int j = 0; j < 4; ++j) *(int4*)(md + j * 4) = pm[j];
    }
    __syncthreads();

    // ---- issue next chunk's global loads; they land during the WMMA work ----
    if (kc + KC < L_)
      issue_chunk_loads(Kp, Vp, Mrow, kc + KC, skey, dseg, pk, pv, pm);

    // ---- load the wave's Q fragments (A layout) from LDS ----
    // A operand, contiguous k order used here: lane holds row m = lane%16; slot i -> k = kbase + i
    Frag aq[4];
    {
      const unsigned short* qb = &Qls[(wave * QW + ln16) * QPITCH];
      const int kbase = hi ? 16 : 0;
#pragma unroll
      for (int s = 0; s < 4; ++s) {
        const unsigned int* lo = (const unsigned int*)(qb + s * 32 + kbase);
        const unsigned int* hi8 = (const unsigned int*)(qb + s * 32 + kbase + 8);
#pragma unroll
        for (int j = 0; j < 4; ++j) { aq[s].d[j] = lo[j]; aq[s].d[4 + j] = hi8[j]; }
      }
    }

    // ---- S = Q * K^T : two 16-key N-tiles, 4 slices over D=128 ----
    // B operand, same contiguous k order as A: lane = column (key); lanes 0-15 hold head-dim 0..15,
    // lanes 16-31 hold 16..31 of each 32-slice, contiguous in the LDS row.
    v8f s0 = {}, s1 = {};
#pragma unroll
    for (int tn = 0; tn < 2; ++tn) {
      const unsigned short* kbp = &Ks[(tn * 16 + ln16) * KPITCH + (hi ? 16 : 0)];
      Frag bk[2];
#pragma unroll
      for (int j = 0; j < 8; ++j) bk[0].d[j] = ((const unsigned int*)kbp)[j];
      v8f c = {};
#pragma unroll
      for (int s = 0; s < 4; ++s) {
        if (s < 3) {
          const unsigned int* src = (const unsigned int*)(kbp + (s + 1) * 32);
#pragma unroll
          for (int j = 0; j < 8; ++j) bk[(s + 1) & 1].d[j] = src[j];
        }
        c = __builtin_amdgcn_wmma_f32_16x16x32_bf16(false, aq[s].v, false, bk[s & 1].v,
                                                    (short)0, c, false, false);
      }
      if (tn == 0) s0 = c; else s1 = c;
    }

    // ---- scale + mask from LDS (mask==0 -> -10000, matching the reference) ----
    // Batch all 8 pair-loads first so the ds_load_2addr_b32's pipeline instead
    // of a wait-0 round trip per row.
    {
      const int mb = (wave * QW + hi * 8) * MPITCH + ln16;
      int m0a[8], m1a[8];
#pragma unroll
      for (int r = 0; r < 8; ++r) {
        m0a[r] = Ms[mb + r * MPITCH];
        m1a[r] = Ms[mb + r * MPITCH + 16];
      }
#pragma unroll
      for (int r = 0; r < 8; ++r) {
        s0[r] = (m0a[r] != 0) ? s0[r] * scale : NEGV;
        s1[r] = (m1a[r] != 0) ? s1[r] * scale : NEGV;
      }
    }

    // ---- online softmax update (row reductions across 16-lane groups) ----
    float p0[8], p1[8], corr[8];
#pragma unroll
    for (int r = 0; r < 8; ++r) {
      float vmax = fmaxf(s0[r], s1[r]);
#pragma unroll
      for (int off = 1; off < 16; off <<= 1)
        vmax = fmaxf(vmax, __shfl_xor(vmax, off, 32));
      const float mnew = fmaxf(mrowv[r], vmax);
      corr[r] = __expf(mrowv[r] - mnew);
      p0[r]   = __expf(s0[r] - mnew);
      p1[r]   = __expf(s1[r] - mnew);
      float rs = p0[r] + p1[r];
#pragma unroll
      for (int off = 1; off < 16; off <<= 1)
        rs += __shfl_xor(rs, off, 32);
      lrow[r] = lrow[r] * corr[r] + rs;
      mrowv[r] = mnew;
    }
#pragma unroll
    for (int t = 0; t < 8; ++t)
#pragma unroll
      for (int r = 0; r < 8; ++r) acc[t][r] *= corr[r];

    // ---- C-layout P -> row-major bf16 in per-wave LDS scratch ----
    {
      unsigned short* pw = &Ps[wave * QW * PPITCH];
#pragma unroll
      for (int r = 0; r < 8; ++r) {
        const int m = r + hi * 8;
        pw[m * PPITCH + ln16]      = f2bf(p0[r]);
        pw[m * PPITCH + 16 + ln16] = f2bf(p1[r]);
      }
    }
    // per-wave LDS RAW: DS ops are in-order within a wave; keep the compiler honest
    asm volatile("s_wait_dscnt 0" ::: "memory");

    // ---- reload P in A-layout ----
    Frag aP;
    {
      const unsigned short* pr = &Ps[wave * QW * PPITCH + ln16 * PPITCH];
      const int kbase = hi ? 16 : 0;
#pragma unroll
      for (int i = 0; i < 16; ++i) {
        const int k = kbase + i;
        aP.u[i] = pr[k];
      }
    }

    // ---- O += P * V : 8 d-tiles; double-buffered B fragments from transposed V ----
    {
      const int kb = hi ? 16 : 0;
      Frag bv[2];
#pragma unroll
      for (int j = 0; j < 8; ++j)
        bv[0].d[j] = ((const unsigned int*)&VT[ln16 * VPITCH + kb])[j];
#pragma unroll
      for (int t = 0; t < 8; ++t) {
        if (t < 7) {
          const unsigned int* src =
              (const unsigned int*)&VT[((t + 1) * 16 + ln16) * VPITCH + kb];
#pragma unroll
          for (int j = 0; j < 8; ++j) bv[(t + 1) & 1].d[j] = src[j];
        }
        acc[t] = __builtin_amdgcn_wmma_f32_16x16x32_bf16(false, aP.v, false, bv[t & 1].v,
                                                         (short)0, acc[t], false, false);
      }
    }
  }

  // ---- epilogue: O = acc / l ----
#pragma unroll
  for (int r = 0; r < 8; ++r) {
    const float inv = 1.0f / lrow[r];
    const int m = q0 + r + hi * 8;
    float* orow = Op + (size_t)m * D_;
#pragma unroll
    for (int t = 0; t < 8; ++t)
      orow[t * 16 + ln16] = acc[t][r] * inv;
  }
}

extern "C" void kernel_launch(void* const* d_in, const int* in_sizes, int n_in,
                              void* d_out, int out_size, void* d_ws, size_t ws_size,
                              hipStream_t stream) {
  const float* Q   = (const float*)d_in[0];
  const float* K   = (const float*)d_in[1];
  const float* V   = (const float*)d_in[2];
  const int*   msk = (const int*)d_in[3];
  float*       O   = (float*)d_out;
  (void)in_sizes; (void)n_in; (void)out_size; (void)d_ws; (void)ws_size;

  dim3 grid(L_ / QB, B_ * H_);   // (32 query-blocks, 32 batch*head)
  dim3 block(128);               // 4 waves (wave32)
  hipLaunchKernelGGL(fa_fwd_bf16_wmma, grid, block, 0, stream, Q, K, V, msk, O);
}
